// MPNEncoder_55791625175040
// MI455X (gfx1250) — matrix operand order repaired and checked by reading, not compiled, not run on hardware
//
#include <hip/hip_runtime.h>
#include <hip/hip_bf16.h>

typedef __attribute__((ext_vector_type(16))) _Float16 v16h;
typedef __attribute__((ext_vector_type(8)))  _Float16 v8h;
typedef __attribute__((ext_vector_type(4)))  _Float16 v4h;
typedef __attribute__((ext_vector_type(8)))  float    v8f;
typedef __attribute__((ext_vector_type(4)))  float    v4f;

#define HID      256
#define N_ATOMS  168000
#define N_BONDS  360000
#define N_MOLS   4000
#define MOL_AT   42
#define ATOM_F   133
#define BOND_F   147
#define KPAD_F   160
#define MAX_NB   6

#define BM 128
#define BN 128
#define NT 8          // 16-wide n-tiles per wave
#define BK 32
#define A_ROW 40      // f16/row (80B): 16 row bases -> disjoint 4-bank ranges
#define C_ROW 68      // f32/row (272B) for epilogue transpose tile

// LDS: 2x A staging buffers (2 * 128*40*2 = 20480B) unioned with
//      C transpose tiles (8 waves * 16*68*4 = 34816B, used twice per wave)
#define SMEM_BYTES 34816
#define ABUF_BYTES (BM * A_ROW * 2)

// ---------------------------------------------------------------------------
// WMMA GEMM:  C[M x 256] = epilogue( A[M x K] @ W.T ),  K = NKT*32
// A is f16, rows padded to lda (zero-padded cols are annihilated by
// zero-padded B columns).
// Bf is fragment-major packed f16: [(kt*16 + ntile)*32 + lane][16],
//   B[k][n] = W[n][k], n = ntile*16 + (lane&15), k = kt*32 + (lane>>4)*16 + e.
// A staged in LDS row-major (lane half h reads contiguous K 16h..16h+15, as in Bf),
// double-buffered; k-loop fully unrolled so all vmem offsets are immediates.
// Epilogue transposes C through LDS (2 passes) so all I/O is 128B vectors.
// ---------------------------------------------------------------------------
template <int NKT, bool HAS_CADD, bool HAS_BIAS,
          bool ST_F32PRE, bool ST_F32RELU, bool ST_F16RELU>
__global__ __launch_bounds__(256)
void gemm_wmma(const _Float16* __restrict__ A, int M, int lda,
               const _Float16* __restrict__ Bf,
               const float* __restrict__ cadd,
               const float* __restrict__ bias,
               float* __restrict__ outF32pre,
               float* __restrict__ outF32relu,
               _Float16* __restrict__ outF16relu)
{
    __shared__ __align__(16) char smem[SMEM_BYTES];
    _Float16* lAf0 = (_Float16*)smem;
    _Float16* lAf1 = (_Float16*)(smem + ABUF_BYTES);

    const int tid  = threadIdx.x;
    const int wave = tid >> 5;
    const int lane = tid & 31;
    const int lhi  = lane >> 4;      // lane half
    const int llo  = lane & 15;
    const int m0   = blockIdx.x * BM;
    const int n0   = blockIdx.y * BN;

    // ---- per-thread loop invariants ---------------------------------------
    int sgrow[2], sdoff[2];
    const _Float16* ssrc[2];
    #pragma unroll
    for (int r = 0; r < 2; ++r) {
        int c    = tid + r * 256;
        int row  = c >> 2;
        int ccol = c & 3;
        sgrow[r] = m0 + row;
        sdoff[r] = row * A_ROW + ccol * 8;
        ssrc[r]  = A + (size_t)sgrow[r] * lda + ccol * 8;
    }
    const int afoff = (wave * 16 + llo) * A_ROW + lhi * 16;
    const _Float16* bbase = Bf + ((size_t)(blockIdx.y * NT * 32 + lane) << 4);

    auto stage = [&](int kt, int buf) {
        _Float16* dst = buf ? lAf1 : lAf0;
        #pragma unroll
        for (int r = 0; r < 2; ++r) {
            v8h hv = v8h{};
            if (sgrow[r] < M)
                hv = *(const v8h*)(ssrc[r] + kt * BK);
            *(v8h*)(dst + sdoff[r]) = hv;
        }
    };

    v8f acc[NT];
    #pragma unroll
    for (int nt = 0; nt < NT; ++nt) acc[nt] = v8f{};

    stage(0, 0);
    #pragma unroll
    for (int kt = 0; kt < NKT; ++kt) {
        __syncthreads();                        // staged tile kt visible
        if (kt + 1 < NKT) stage(kt + 1, (kt + 1) & 1);  // prefetch next tile

        const _Float16* lbuf = (kt & 1) ? lAf1 : lAf0;
        v8h a0 = *(const v8h*)(lbuf + afoff);
        v8h a1 = *(const v8h*)(lbuf + afoff + 8);
        v16h afrag = __builtin_shufflevector(a0, a1,
            0, 1, 2, 3, 4, 5, 6, 7, 8, 9, 10, 11, 12, 13, 14, 15);

        #pragma unroll
        for (int nt = 0; nt < NT; ++nt) {
            const _Float16* bp = bbase + kt * 8192 + nt * 512;  // imm offsets
            v8h b0 = *(const v8h*)bp;
            v8h b1 = *(const v8h*)(bp + 8);
            v16h bfrag = __builtin_shufflevector(b0, b1,
                0, 1, 2, 3, 4, 5, 6, 7, 8, 9, 10, 11, 12, 13, 14, 15);
            acc[nt] = __builtin_amdgcn_wmma_f32_16x16x32_f16(
                false, afrag, false, bfrag, (short)0, acc[nt], false, false);
        }
    }

    // ---- epilogue: transpose C tile through LDS, then 128B vector I/O -----
    __syncthreads();                       // staging buffers -> C tiles
    float* lC = (float*)smem + wave * (16 * C_ROW);
    const int colb = llo * 4;              // 0..60, 16B aligned
    #pragma unroll
    for (int p = 0; p < 2; ++p) {          // two 64-col passes (private region;
        #pragma unroll                     //  same-wave DS ordering is in-order)
        for (int t = 0; t < 4; ++t) {
            int col = t * 16 + llo;
            #pragma unroll
            for (int r = 0; r < 8; ++r)
                lC[(r + 8 * lhi) * C_ROW + col] = acc[p * 4 + t][r];
        }
        const int gcol = n0 + p * 64 + colb;
        v4f bvec;
        if constexpr (HAS_BIAS) bvec = *(const v4f*)(bias + gcol);
        #pragma unroll
        for (int i = 0; i < 8; ++i) {
            int row  = i * 2 + lhi;
            int grow = m0 + wave * 16 + row;
            if (grow < M) {
                v4f v = *(const v4f*)&lC[row * C_ROW + colb];
                if constexpr (HAS_BIAS) {
                    #pragma unroll
                    for (int e = 0; e < 4; ++e) v[e] += bvec[e];
                }
                if constexpr (HAS_CADD) {
                    v4f cv = *(const v4f*)(cadd + (size_t)grow * HID + gcol);
                    #pragma unroll
                    for (int e = 0; e < 4; ++e) v[e] += cv[e];
                }
                if constexpr (ST_F32PRE)
                    *(v4f*)(outF32pre + (size_t)grow * HID + gcol) = v;
                v4f vr;
                #pragma unroll
                for (int e = 0; e < 4; ++e) vr[e] = v[e] > 0.f ? v[e] : 0.f;
                if constexpr (ST_F32RELU)
                    *(v4f*)(outF32relu + (size_t)grow * HID + gcol) = vr;
                if constexpr (ST_F16RELU) {
                    v4h hv;
                    #pragma unroll
                    for (int e = 0; e < 4; ++e) hv[e] = (_Float16)vr[e];
                    *(v4h*)(outF16relu + (size_t)grow * HID + gcol) = hv;
                }
            }
        }
    }
}

// ---------------------------------------------------------------------------
// Convert f32 [M x Kreal] -> f16 [M x 160], zero-padded columns.
// One thread per 8-col chunk (20 chunks per row).
// ---------------------------------------------------------------------------
__global__ __launch_bounds__(256)
void convert_pad(const float* __restrict__ src, _Float16* __restrict__ dst,
                 int M, int Kreal)
{
    int idx = blockIdx.x * 256 + threadIdx.x;
    int row = idx / 20;
    int j   = idx - row * 20;          // chunk of 8 cols
    if (row >= M) return;
    v8h hv = v8h{};
    #pragma unroll
    for (int e = 0; e < 8; ++e) {
        int k = j * 8 + e;
        if (k < Kreal) hv[e] = (_Float16)src[(size_t)row * Kreal + k];
    }
    *(v8h*)(dst + (size_t)row * KPAD_F + j * 8) = hv;
}

// ---------------------------------------------------------------------------
// Pack weights into fragment-major f16 layout (see gemm_wmma header comment).
// ---------------------------------------------------------------------------
__global__ __launch_bounds__(256)
void pack_weights(const float* __restrict__ Wi, const float* __restrict__ Wh,
                  const float* __restrict__ Wo,
                  _Float16* __restrict__ Bi, _Float16* __restrict__ Bh,
                  _Float16* __restrict__ BoA, _Float16* __restrict__ BoM)
{
    int idx = blockIdx.x * 256 + threadIdx.x;   // up to 65536
    int e    = idx & 15;
    int lane = (idx >> 4) & 31;
    int t    = idx >> 9;
    int ntg  = t & 15;
    int kt   = t >> 4;
    int k    = kt * 32 + ((lane >> 4) << 4) + e;
    int n    = ntg * 16 + (lane & 15);

    if (idx < 5 * 16 * 512) {   // K = 160 tiles
        Bi[idx]  = (_Float16)(k < BOND_F ? Wi[n * BOND_F + k] : 0.f);
        BoA[idx] = (_Float16)(k < ATOM_F ? Wo[n * (ATOM_F + HID) + k] : 0.f);
    }
    if (idx < 8 * 16 * 512) {   // K = 256 tiles
        Bh[idx]  = (_Float16)Wh[n * HID + k];
        BoM[idx] = (_Float16)Wo[n * (ATOM_F + HID) + ATOM_F + k];
    }
}

// ---------------------------------------------------------------------------
// a_msg[a][:] = sum_{j<6} msg[a2b[a][j]][:]   (one wave per atom, 8 f16/lane)
// ---------------------------------------------------------------------------
__global__ __launch_bounds__(256)
void gather_sum(const _Float16* __restrict__ msg, const int* __restrict__ a2b,
                _Float16* __restrict__ amsg, int n_rows)
{
    int gw   = (blockIdx.x * 256 + threadIdx.x) >> 5;
    int lane = threadIdx.x & 31;
    if (gw >= n_rows) return;
    float accv[8] = {0.f, 0.f, 0.f, 0.f, 0.f, 0.f, 0.f, 0.f};
    #pragma unroll
    for (int j = 0; j < MAX_NB; ++j) {
        int b = a2b[gw * MAX_NB + j];
        v8h x = *(const v8h*)(msg + (size_t)b * HID + lane * 8);
        #pragma unroll
        for (int e = 0; e < 8; ++e) accv[e] += (float)x[e];
    }
    v8h o;
    #pragma unroll
    for (int e = 0; e < 8; ++e) o[e] = (_Float16)accv[e];
    *(v8h*)(amsg + (size_t)gw * HID + lane * 8) = o;
}

// ---------------------------------------------------------------------------
// tmp[b][:] = amsg[b2a[b]][:] - msg[b2revb[b]][:]
// ---------------------------------------------------------------------------
__global__ __launch_bounds__(256)
void bond_update(const _Float16* __restrict__ amsg, const _Float16* __restrict__ msg,
                 const int* __restrict__ b2a, const int* __restrict__ b2revb,
                 _Float16* __restrict__ tmp, int n_rows)
{
    int gw   = (blockIdx.x * 256 + threadIdx.x) >> 5;
    int lane = threadIdx.x & 31;
    if (gw >= n_rows) return;
    int a  = b2a[gw];
    int rb = b2revb[gw];
    v8h x = *(const v8h*)(amsg + (size_t)a  * HID + lane * 8);
    v8h y = *(const v8h*)(msg  + (size_t)rb * HID + lane * 8);
    v8h o;
    #pragma unroll
    for (int e = 0; e < 8; ++e) o[e] = (_Float16)((float)x[e] - (float)y[e]);
    *(v8h*)(tmp + (size_t)gw * HID + lane * 8) = o;
}

// ---------------------------------------------------------------------------
// mol_vecs[mol][h] = mean over 42 atoms of dfin
// ---------------------------------------------------------------------------
__global__ __launch_bounds__(256)
void mol_mean(const float* __restrict__ dfin, float* __restrict__ out)
{
    int mol = blockIdx.x;
    int h   = threadIdx.x;
    float s = 0.f;
    #pragma unroll 6
    for (int a = 0; a < MOL_AT; ++a)
        s += dfin[((size_t)mol * MOL_AT + a) * HID + h];
    out[(size_t)mol * HID + h] = s * (1.f / (float)MOL_AT);
}

// ---------------------------------------------------------------------------
extern "C" void kernel_launch(void* const* d_in, const int* in_sizes, int n_in,
                              void* d_out, int out_size, void* d_ws, size_t ws_size,
                              hipStream_t stream)
{
    (void)in_sizes; (void)n_in; (void)out_size; (void)ws_size;

    const float* f_atoms = (const float*)d_in[0];
    const float* f_bonds = (const float*)d_in[1];
    const float* W_i     = (const float*)d_in[2];
    const float* W_h     = (const float*)d_in[3];
    const float* W_o     = (const float*)d_in[4];
    const float* b_o     = (const float*)d_in[5];
    const int*   a2b     = (const int*)d_in[6];
    const int*   b2a     = (const int*)d_in[7];
    const int*   b2revb  = (const int*)d_in[8];

    float* out = (float*)d_out;
    float* d0   = out;
    float* d1   = out + (size_t)N_ATOMS * HID;
    float* d2   = out + (size_t)N_ATOMS * HID * 2;
    float* dfin = out + (size_t)N_ATOMS * HID * 3;
    float* mv   = out + (size_t)N_ATOMS * HID * 4;

    // ---- workspace carve --------------------------------------------------
    char* w = (char*)d_ws;
    float*    inp  = (float*)w;    w += (size_t)N_BONDS * HID * 4;   // pre-relu W_i out
    float*    base = (float*)w;    w += (size_t)N_ATOMS * HID * 4;   // f_atoms@Wo_a + b
    _Float16* msg  = (_Float16*)w; w += (size_t)N_BONDS * HID * 2;   // relu msgs (f16)
    _Float16* amsg = (_Float16*)w; w += (size_t)N_ATOMS * HID * 2;   // gathered (f16)
    _Float16* tmp  = (_Float16*)w; w += (size_t)N_BONDS * HID * 2;   // bond diff (f16)
    _Float16* fb16 = (_Float16*)w; w += (size_t)N_BONDS * KPAD_F * 2; // f_bonds f16 pad
    _Float16* fa16 = (_Float16*)w; w += (size_t)N_ATOMS * KPAD_F * 2; // f_atoms f16 pad
    _Float16* Bi   = (_Float16*)w; w += (size_t)KPAD_F * HID * 2;    // frag-packed
    _Float16* Bh   = (_Float16*)w; w += (size_t)HID * HID * 2;
    _Float16* BoA  = (_Float16*)w; w += (size_t)KPAD_F * HID * 2;
    _Float16* BoM  = (_Float16*)w; w += (size_t)HID * HID * 2;

    const dim3 blk(256);
    const dim3 gB((N_BONDS + BM - 1) / BM, HID / BN);   // 2813 x 2
    const dim3 gA((N_ATOMS + BM - 1) / BM, HID / BN);   // 1313 x 2
    const int  gwB = (N_BONDS * 32 + 255) / 256;        // wave-per-bond grids
    const int  gwA = (N_ATOMS * 32 + 255) / 256;

    // 0) pack weights + convert features to padded f16
    pack_weights<<<(HID * HID + 255) / 256, blk, 0, stream>>>(W_i, W_h, W_o, Bi, Bh, BoA, BoM);
    convert_pad<<<(N_BONDS * 20 + 255) / 256, blk, 0, stream>>>(f_bonds, fb16, N_BONDS, BOND_F);
    convert_pad<<<(N_ATOMS * 20 + 255) / 256, blk, 0, stream>>>(f_atoms, fa16, N_ATOMS, ATOM_F);

    // 1) inp = f_bonds @ W_i.T ; msg = relu(inp)  (f16)
    gemm_wmma<5, false, false, true, false, true>
        <<<gB, blk, 0, stream>>>(fb16, N_BONDS, KPAD_F,
                                 Bi, nullptr, nullptr, inp, nullptr, msg);

    // 2) base = f_atoms @ Wo_atom.T + b_o ; d0 = relu(base)
    gemm_wmma<5, false, true, true, true, false>
        <<<gA, blk, 0, stream>>>(fa16, N_ATOMS, KPAD_F,
                                 BoA, nullptr, b_o, base, d0, nullptr);

    // ---- depth iterations -------------------------------------------------
    float* dsave[3] = {d1, d2, dfin};
    for (int d = 0; d < 3; ++d) {
        // gather: amsg = sum_nb msg[a2b]
        gather_sum<<<gwA, blk, 0, stream>>>(msg, a2b, amsg, N_ATOMS);

        // atom hidden: dX = relu(base + amsg @ Wo_msg.T)
        gemm_wmma<8, true, false, false, true, false>
            <<<gA, blk, 0, stream>>>(amsg, N_ATOMS, HID,
                                     BoM, base, nullptr, nullptr, dsave[d], nullptr);

        if (d < 2) {
            // tmp = amsg[b2a] - msg[b2revb]
            bond_update<<<gwB, blk, 0, stream>>>(amsg, msg, b2a, b2revb, tmp, N_BONDS);
            // msg = relu(inp + tmp @ W_h.T)   (f16)
            gemm_wmma<8, true, false, false, false, true>
                <<<gB, blk, 0, stream>>>(tmp, N_BONDS, HID,
                                         Bh, inp, nullptr, nullptr, nullptr, msg);
        }
    }

    // mol_vecs = mean over atoms of dfin
    mol_mean<<<N_MOLS, blk, 0, stream>>>(dfin, mv);
}
